// MNIST_CNN2D_TWN_42597485641965
// MI455X (gfx1250) — matrix operand order repaired and checked by reading, not compiled, not run on hardware
//
#include <hip/hip_runtime.h>

// ---------------------------------------------------------------------------
// MNIST TWN CNN forward on gfx1250 (CDNA5): bf16 WMMA for conv1/conv2/fc1.
// B operands pre-swizzled to [K/32][N][32] so LDS tiles are column-major and
// every WMMA fragment is fetched with ds_load_b128 (no scalar u16 gathers).
// ---------------------------------------------------------------------------

typedef __attribute__((ext_vector_type(16))) __bf16 v16bf;
typedef __attribute__((ext_vector_type(8)))  float  v8f;

#define BN_EPS 1e-5f
#define TWN_T  0.05f
#define LDA 40   // padded LDS stride (elements): 80B keeps b128 align, kills bank conflicts
#define LDB 40

// ---------------- WMMA helpers (layouts per CDNA5 ISA 7.12.2) ---------------

// A-matrix 16x32 bf16, LDS row-major [row][k] (stride LDA):
// lane (m = lane%16, h = lane/16): a[i] = A[m][8h+i], a[8+i] = A[m][16+8h+i]
// -> two contiguous 8-element runs -> 2x ds_load_b128
__device__ inline v16bf pack_A(const __bf16* As, int row, int h) {
  v16bf a;
#pragma unroll
  for (int i = 0; i < 8; ++i) {
    a[i]     = As[row * LDA + 8 * h + i];
    a[i + 8] = As[row * LDA + 16 + 8 * h + i];
  }
  return a;
}

// B-matrix 32x16 bf16, LDS column-major [n][k] (stride LDB):
// lane (n = col, h): b[i] = B[8h+i][n], b[8+i] = B[16+8h+i][n]
// -> two contiguous 8-element runs -> 2x ds_load_b128
__device__ inline v16bf pack_Bcm(const __bf16* Bs, int col, int h) {
  v16bf b;
#pragma unroll
  for (int i = 0; i < 8; ++i) {
    b[i]     = Bs[col * LDB + 8 * h + i];
    b[i + 8] = Bs[col * LDB + 16 + 8 * h + i];
  }
  return b;
}

__device__ inline v8f wmma_bf16(v16bf a, v16bf b, v8f c) {
  return __builtin_amdgcn_wmma_f32_16x16x32_bf16(
      /*neg_a=*/false, a, /*neg_b=*/false, b,
      /*c_mod=*/(short)0, c, /*reuse_a=*/false, /*reuse_b=*/false);
}

union bf2u { __bf16 h[2]; unsigned int u; };

// ---------------- small utility kernels -------------------------------------

__global__ void k_zero_u32(unsigned int* p, int n) {
  int i = blockIdx.x * blockDim.x + threadIdx.x;
  if (i < n) p[i] = 0u;
}

__global__ void k_absmax(const float* __restrict__ w, int n, unsigned int* slot) {
  __shared__ unsigned int sm[256];
  unsigned int v = 0u;
  for (int i = blockIdx.x * blockDim.x + threadIdx.x; i < n;
       i += gridDim.x * blockDim.x) {
    unsigned int u = __float_as_uint(fabsf(w[i]));
    v = (u > v) ? u : v;
  }
  sm[threadIdx.x] = v;
  __syncthreads();
  for (int st = 128; st > 0; st >>= 1) {
    if (threadIdx.x < st) {
      unsigned int o = sm[threadIdx.x + st];
      if (o > sm[threadIdx.x]) sm[threadIdx.x] = o;
    }
    __syncthreads();
  }
  if (threadIdx.x == 0) atomicMax(slot, sm[0]);   // order-independent
}

__device__ inline float tern(float w, float t) {
  return (w > t) ? 1.f : ((w < -t) ? -1.f : 0.f);
}

// swizzled quant: q[((k>>5)*N + n)*32 + (k&31)] = k<Ksrc ? tern(w[n*Ksrc+k]) : 0
__global__ void k_quant_T(const float* __restrict__ w, const float* __restrict__ amax,
                          __bf16* __restrict__ q, int Kd, int N, int Ksrc) {
  int i = blockIdx.x * blockDim.x + threadIdx.x;
  if (i >= Kd * N) return;
  float t = TWN_T * amax[0];
  int k = i / N, n = i % N;
  float v = (k < Ksrc) ? tern(w[n * Ksrc + k], t) : 0.f;
  q[(((size_t)(k >> 5) * N + n) << 5) + (k & 31)] = (__bf16)v;
}

__global__ void k_quant_f32(const float* __restrict__ w, const float* __restrict__ amax,
                            float* __restrict__ q, int n) {
  int i = blockIdx.x * blockDim.x + threadIdx.x;
  if (i < n) q[i] = tern(w[i], TWN_T * amax[0]);
}

// per-channel mean & biased variance; one block per channel, fixed order
__global__ void k_bn_stats(const float* __restrict__ x, float* __restrict__ mean,
                           float* __restrict__ var, int C, int B, int HW) {
  __shared__ float ss[256], sq[256];
  int c = blockIdx.x, tid = threadIdx.x;
  long N = (long)B * HW;
  float s = 0.f, s2 = 0.f;
  for (long j = tid; j < N; j += 256) {
    int b = (int)(j / HW), i = (int)(j % HW);
    float v = x[((long)b * C + c) * HW + i];
    s += v;
    s2 += v * v;
  }
  ss[tid] = s; sq[tid] = s2;
  __syncthreads();
  for (int st = 128; st > 0; st >>= 1) {
    if (tid < st) { ss[tid] += ss[tid + st]; sq[tid] += sq[tid + st]; }
    __syncthreads();
  }
  if (tid == 0) {
    float m = ss[0] / (float)N;
    mean[c] = m;
    var[c] = sq[0] / (float)N - m * m;
  }
}

// BN (train) + ReLU + 2x2 maxpool, f32 out.  out flat == [B][C][H/2][W/2]
__global__ void k_bn_relu_pool_f32(const float* __restrict__ raw,
                                   const float* __restrict__ mean,
                                   const float* __restrict__ var,
                                   const float* __restrict__ g,
                                   const float* __restrict__ be,
                                   float* __restrict__ outp,
                                   int C, int H, int W, int total) {
  int i = blockIdx.x * blockDim.x + threadIdx.x;
  if (i >= total) return;
  int W2 = W >> 1, H2 = H >> 1;
  int x2 = i % W2; int t = i / W2;
  int y2 = t % H2; t /= H2;
  int c = t % C;   int b = t / C;
  float m = mean[c], rs = rsqrtf(var[c] + BN_EPS), ga = g[c], bb = be[c];
  const float* p = raw + (((long)b * C + c) * H + 2 * y2) * W + 2 * x2;
  float best = -3.4e38f;
#pragma unroll
  for (int dy = 0; dy < 2; ++dy)
#pragma unroll
    for (int dx = 0; dx < 2; ++dx) {
      float v = (p[dy * W + dx] - m) * rs * ga + bb;
      v = v > 0.f ? v : 0.f;
      best = v > best ? v : best;
    }
  outp[i] = best;
}

// same, bf16 out (feeds fc1's A operand; flat order matches reshape(B,-1))
__global__ void k_bn_relu_pool_bf16(const float* __restrict__ raw,
                                    const float* __restrict__ mean,
                                    const float* __restrict__ var,
                                    const float* __restrict__ g,
                                    const float* __restrict__ be,
                                    __bf16* __restrict__ outp,
                                    int C, int H, int W, int total) {
  int i = blockIdx.x * blockDim.x + threadIdx.x;
  if (i >= total) return;
  int W2 = W >> 1, H2 = H >> 1;
  int x2 = i % W2; int t = i / W2;
  int y2 = t % H2; t /= H2;
  int c = t % C;   int b = t / C;
  float m = mean[c], rs = rsqrtf(var[c] + BN_EPS), ga = g[c], bb = be[c];
  const float* p = raw + (((long)b * C + c) * H + 2 * y2) * W + 2 * x2;
  float best = -3.4e38f;
#pragma unroll
  for (int dy = 0; dy < 2; ++dy)
#pragma unroll
    for (int dx = 0; dx < 2; ++dx) {
      float v = (p[dy * W + dx] - m) * rs * ga + bb;
      v = v > 0.f ? v : 0.f;
      best = v > best ? v : best;
    }
  outp[i] = (__bf16)best;
}

__global__ void k_bn_relu_f32(const float* __restrict__ raw,
                              const float* __restrict__ mean,
                              const float* __restrict__ var,
                              const float* __restrict__ g,
                              const float* __restrict__ be,
                              float* __restrict__ outp, int C, int total) {
  int i = blockIdx.x * blockDim.x + threadIdx.x;
  if (i >= total) return;
  int c = i % C;
  float v = (raw[i] - mean[c]) * rsqrtf(var[c] + BN_EPS) * g[c] + be[c];
  outp[i] = v > 0.f ? v : 0.f;
}

// fc2: [512,512] x [10,512]^T + b  (tiny, plain f32)
__global__ void k_fc2(const float* __restrict__ h, const float* __restrict__ q,
                      const float* __restrict__ bias, float* __restrict__ outp) {
  int i = blockIdx.x * blockDim.x + threadIdx.x;
  if (i >= 512 * 10) return;
  int b = i / 10, j = i % 10;
  const float* hr = h + b * 512;
  const float* qr = q + j * 512;
  float s = bias[j];
  for (int c = 0; c < 512; ++c) s += hr[c] * qr[c];
  outp[i] = s;
}

// ---------------- WMMA GEMM kernels -----------------------------------------

// conv1 implicit GEMM: M=512*576, N=64, K=25 (padded to 32); tile 64x64 per WG
__global__ void __launch_bounds__(128)
k_conv1_wmma(const float* __restrict__ x, const __bf16* __restrict__ q1,
             const float* __restrict__ bias, float* __restrict__ raw1) {
  __shared__ __bf16 As[64 * LDA];   // [row][k]
  __shared__ __bf16 Bs[64 * LDB];   // [oc][k] column-major
  int tid = threadIdx.x;
  int wave = tid >> 5, lane = tid & 31;
  int mrow = lane & 15, h = lane >> 4;
  int b = blockIdx.x / 9;               // 576/64 = 9 row-tiles per image
  int sp0 = (blockIdx.x % 9) * 64;

  // A tile: im2col, pair-packed b32 stores
  for (int t = tid; t < 1024; t += 128) {
    int r = t >> 4, kp = t & 15;
    int sp = sp0 + r;
    int oy = sp / 24, ox = sp % 24;
    float v0 = 0.f, v1 = 0.f;
    int k = 2 * kp;
    if (k < 25) {
      int ky = k / 5, kx = k % 5;
      v0 = x[b * 784 + (oy + ky) * 28 + (ox + kx)];
    }
    if (k + 1 < 25) {
      int ky = (k + 1) / 5, kx = (k + 1) % 5;
      v1 = x[b * 784 + (oy + ky) * 28 + (ox + kx)];
    }
    bf2u p; p.h[0] = (__bf16)v0; p.h[1] = (__bf16)v1;
    *(unsigned int*)(As + r * LDA + 2 * kp) = p.u;
  }
  // B tile: contiguous [64][32] -> column-major padded LDS, b128 copies
  {
    const uint4* s4 = (const uint4*)q1;
    for (int c = tid; c < 64 * 4; c += 128) {
      int n = c >> 2, part = c & 3;
      *(uint4*)(Bs + n * LDB + part * 8) = s4[c];
    }
  }
  __syncthreads();

  v8f z = {0.f, 0.f, 0.f, 0.f, 0.f, 0.f, 0.f, 0.f};
  v8f acc[4];
#pragma unroll
  for (int j = 0; j < 4; ++j) acc[j] = z;

  v16bf a = pack_A(As, wave * 16 + mrow, h);
#pragma unroll
  for (int j = 0; j < 4; ++j) {
    v16bf bb = pack_Bcm(Bs, j * 16 + mrow, h);
    acc[j] = wmma_bf16(a, bb, acc[j]);
  }

#pragma unroll
  for (int j = 0; j < 4; ++j) {
    int oc = j * 16 + mrow;
    float bi = bias[oc];
#pragma unroll
    for (int r = 0; r < 8; ++r) {
      int sp = sp0 + wave * 16 + r + 8 * h;
      int oy = sp / 24, ox = sp % 24;
      raw1[(((long)b * 64 + oc) * 24 + oy) * 24 + ox] = acc[j][r] + bi;
    }
  }
}

// conv2 implicit GEMM: per-WG one image: M=64 (8x8), N=128, K=1600 in 50 steps
__global__ void __launch_bounds__(128)
k_conv2_wmma(const float* __restrict__ pooled1, const __bf16* __restrict__ q2,
             const float* __restrict__ bias, float* __restrict__ raw2) {
  __shared__ __bf16 As[64 * LDA];    // [sp][k]
  __shared__ __bf16 Bs[128 * LDB];   // [oc][k] column-major
  int tid = threadIdx.x;
  int wave = tid >> 5, lane = tid & 31;
  int mrow = lane & 15, h = lane >> 4;
  int b = blockIdx.x;
  const float* inb = pooled1 + (long)b * 64 * 144;

  v8f z = {0.f, 0.f, 0.f, 0.f, 0.f, 0.f, 0.f, 0.f};
  v8f acc[8];
#pragma unroll
  for (int j = 0; j < 8; ++j) acc[j] = z;

  for (int k0 = 0; k0 < 1600; k0 += 32) {
    // A tile: im2col, pair-packed b32 stores
    for (int t = tid; t < 1024; t += 128) {
      int r = t >> 4, kp = t & 15;
      int oy = r >> 3, ox = r & 7;
      int k = k0 + 2 * kp;
      int ic0 = k / 25, rem0 = k % 25;
      int ic1 = (k + 1) / 25, rem1 = (k + 1) % 25;
      float v0 = inb[ic0 * 144 + (oy + rem0 / 5) * 12 + (ox + rem0 % 5)];
      float v1 = inb[ic1 * 144 + (oy + rem1 / 5) * 12 + (ox + rem1 % 5)];
      bf2u p; p.h[0] = (__bf16)v0; p.h[1] = (__bf16)v1;
      *(unsigned int*)(As + r * LDA + 2 * kp) = p.u;
    }
    // B tile: swizzled global layout [kt][128][32] is one contiguous 8KB block
    {
      const uint4* s4 = (const uint4*)(q2 + (size_t)k0 * 128);
      for (int c = tid; c < 128 * 4; c += 128) {
        int n = c >> 2, part = c & 3;
        *(uint4*)(Bs + n * LDB + part * 8) = s4[c];
      }
      if (k0 + 32 < 1600)
        __builtin_prefetch(q2 + (size_t)(k0 + 32) * 128 + (tid << 5), 0, 1);
    }
    __syncthreads();

    v16bf a = pack_A(As, wave * 16 + mrow, h);
#pragma unroll
    for (int j = 0; j < 8; ++j) {
      v16bf bb = pack_Bcm(Bs, j * 16 + mrow, h);
      acc[j] = wmma_bf16(a, bb, acc[j]);
    }
    __syncthreads();
  }

#pragma unroll
  for (int j = 0; j < 8; ++j) {
    int oc = j * 16 + mrow;
    float bi = bias[oc];
#pragma unroll
    for (int r = 0; r < 8; ++r) {
      int sp = wave * 16 + r + 8 * h;
      int oy = sp >> 3, ox = sp & 7;
      raw2[(((long)b * 128 + oc) * 8 + oy) * 8 + ox] = acc[j][r] + bi;
    }
  }
}

// fc1 GEMM: M=512, N=512, K=2048; tile 64x128 per WG, grid (8,4)
__global__ void __launch_bounds__(128)
k_fc1_wmma(const __bf16* __restrict__ A, const __bf16* __restrict__ Bq,
           const float* __restrict__ bias, float* __restrict__ raw3) {
  __shared__ __bf16 As[64 * LDA];
  __shared__ __bf16 Bs[128 * LDB];
  int tid = threadIdx.x;
  int wave = tid >> 5, lane = tid & 31;
  int mrow = lane & 15, h = lane >> 4;
  int m0 = blockIdx.x * 64, n0 = blockIdx.y * 128;

  v8f z = {0.f, 0.f, 0.f, 0.f, 0.f, 0.f, 0.f, 0.f};
  v8f acc[8];
#pragma unroll
  for (int j = 0; j < 8; ++j) acc[j] = z;

  for (int k0 = 0; k0 < 2048; k0 += 32) {
    // A tile: row-major 64x32, b128 copies
    for (int c = tid; c < 256; c += 128) {
      int r = c >> 2, part = c & 3;
      *(uint4*)(As + r * LDA + part * 8) =
          *(const uint4*)(A + (size_t)(m0 + r) * 2048 + k0 + part * 8);
    }
    // B tile: swizzled layout -> contiguous 128*32 block at k0*512 + n0*32
    {
      const uint4* s4 = (const uint4*)(Bq + (size_t)k0 * 512 + (size_t)n0 * 32);
      for (int c = tid; c < 128 * 4; c += 128) {
        int n = c >> 2, part = c & 3;
        *(uint4*)(Bs + n * LDB + part * 8) = s4[c];
      }
      if (k0 + 32 < 2048)
        __builtin_prefetch(Bq + (size_t)(k0 + 32) * 512 + (size_t)n0 * 32 + (tid << 5), 0, 1);
    }
    __syncthreads();

    v16bf a = pack_A(As, wave * 16 + mrow, h);
#pragma unroll
    for (int j = 0; j < 8; ++j) {
      v16bf bb = pack_Bcm(Bs, j * 16 + mrow, h);
      acc[j] = wmma_bf16(a, bb, acc[j]);
    }
    __syncthreads();
  }

#pragma unroll
  for (int j = 0; j < 8; ++j) {
    int n = n0 + j * 16 + mrow;
    float bi = bias[n];
#pragma unroll
    for (int r = 0; r < 8; ++r) {
      int mm = m0 + wave * 16 + r + 8 * h;
      raw3[(long)mm * 512 + n] = acc[j][r] + bi;
    }
  }
}

// ---------------- host-side orchestration -----------------------------------

extern "C" void kernel_launch(void* const* d_in, const int* in_sizes, int n_in,
                              void* d_out, int out_size, void* d_ws, size_t ws_size,
                              hipStream_t stream) {
  (void)in_sizes; (void)n_in; (void)out_size; (void)ws_size;

  const float* x       = (const float*)d_in[0];
  const float* conv1_w = (const float*)d_in[1];
  const float* conv1_b = (const float*)d_in[2];
  const float* bn1_g   = (const float*)d_in[3];
  const float* bn1_be  = (const float*)d_in[4];
  const float* conv2_w = (const float*)d_in[5];
  const float* conv2_b = (const float*)d_in[6];
  const float* bn2_g   = (const float*)d_in[7];
  const float* bn2_be  = (const float*)d_in[8];
  const float* fc1_w   = (const float*)d_in[9];
  const float* fc1_b   = (const float*)d_in[10];
  const float* bn3_g   = (const float*)d_in[11];
  const float* bn3_be  = (const float*)d_in[12];
  const float* fc2_w   = (const float*)d_in[13];
  const float* fc2_b   = (const float*)d_in[14];
  float* out = (float*)d_out;

  // bump allocator over workspace (256-B aligned slots)
  char* base = (char*)d_ws;
  size_t off = 0;
  auto take = [&](size_t bytes) -> char* {
    char* p = base + off;
    off = (off + bytes + 255) & ~(size_t)255;
    return p;
  };
  float*  absm  = (float*) take(4 * 4);
  float*  mean1 = (float*) take(64 * 4);
  float*  var1  = (float*) take(64 * 4);
  float*  mean2 = (float*) take(128 * 4);
  float*  var2  = (float*) take(128 * 4);
  float*  mean3 = (float*) take(512 * 4);
  float*  var3  = (float*) take(512 * 4);
  __bf16* q1    = (__bf16*)take((size_t)32 * 64 * 2);
  __bf16* q2    = (__bf16*)take((size_t)1600 * 128 * 2);
  __bf16* qf1   = (__bf16*)take((size_t)2048 * 512 * 2);
  float*  qf2   = (float*) take((size_t)10 * 512 * 4);
  float*  raw1  = (float*) take((size_t)512 * 64 * 576 * 4);   // 75.5 MB
  float*  pool1 = (float*) take((size_t)512 * 64 * 144 * 4);   // 18.9 MB
  float*  raw2  = (float*) take((size_t)512 * 128 * 64 * 4);   // 16.8 MB
  __bf16* act2  = (__bf16*)take((size_t)512 * 2048 * 2);       //  2.1 MB
  float*  raw3  = (float*) take((size_t)512 * 512 * 4);
  float*  h3    = (float*) take((size_t)512 * 512 * 4);

  // 1) ternary quantization (absmax -> threshold -> {-1,0,+1} in bf16)
  k_zero_u32<<<1, 32, 0, stream>>>((unsigned int*)absm, 4);
  k_absmax<<<(1600 + 255) / 256, 256, 0, stream>>>(conv1_w, 64 * 25,
                                                   (unsigned int*)&absm[0]);
  k_absmax<<<(204800 + 255) / 256, 256, 0, stream>>>(conv2_w, 128 * 1600,
                                                     (unsigned int*)&absm[1]);
  k_absmax<<<(1048576 + 255) / 256, 256, 0, stream>>>(fc1_w, 512 * 2048,
                                                      (unsigned int*)&absm[2]);
  k_absmax<<<(5120 + 255) / 256, 256, 0, stream>>>(fc2_w, 10 * 512,
                                                   (unsigned int*)&absm[3]);
  k_quant_T<<<(32 * 64 + 255) / 256, 256, 0, stream>>>(conv1_w, &absm[0], q1, 32, 64, 25);
  k_quant_T<<<(1600 * 128 + 255) / 256, 256, 0, stream>>>(conv2_w, &absm[1], q2, 1600, 128, 1600);
  k_quant_T<<<(2048 * 512 + 255) / 256, 256, 0, stream>>>(fc1_w, &absm[2], qf1, 2048, 512, 2048);
  k_quant_f32<<<(5120 + 255) / 256, 256, 0, stream>>>(fc2_w, &absm[3], qf2, 5120);

  // 2) conv1 (WMMA) -> BN1 stats -> BN+ReLU+pool
  k_conv1_wmma<<<4608, 128, 0, stream>>>(x, q1, conv1_b, raw1);
  k_bn_stats<<<64, 256, 0, stream>>>(raw1, mean1, var1, 64, 512, 576);
  {
    int total = 512 * 64 * 144;
    k_bn_relu_pool_f32<<<(total + 255) / 256, 256, 0, stream>>>(
        raw1, mean1, var1, bn1_g, bn1_be, pool1, 64, 24, 24, total);
  }

  // 3) conv2 (WMMA) -> BN2 stats -> BN+ReLU+pool (bf16 activations)
  k_conv2_wmma<<<512, 128, 0, stream>>>(pool1, q2, conv2_b, raw2);
  k_bn_stats<<<128, 256, 0, stream>>>(raw2, mean2, var2, 128, 512, 64);
  {
    int total = 512 * 128 * 16;
    k_bn_relu_pool_bf16<<<(total + 255) / 256, 256, 0, stream>>>(
        raw2, mean2, var2, bn2_g, bn2_be, act2, 128, 8, 8, total);
  }

  // 4) fc1 (WMMA) -> BN3 stats -> BN+ReLU -> fc2
  k_fc1_wmma<<<dim3(8, 4), 128, 0, stream>>>(act2, qf1, fc1_b, raw3);
  k_bn_stats<<<512, 256, 0, stream>>>(raw3, mean3, var3, 512, 512, 1);
  {
    int total = 512 * 512;
    k_bn_relu_f32<<<(total + 255) / 256, 256, 0, stream>>>(
        raw3, mean3, var3, bn3_g, bn3_be, h3, 512, total);
  }
  k_fc2<<<(5120 + 255) / 256, 256, 0, stream>>>(h3, qf2, fc2_b, out);
}
